// WanSelfAttention_86096914415843
// MI455X (gfx1250) — matrix operand order repaired and checked by reading, not compiled, not run on hardware
//
#include <hip/hip_runtime.h>
#include <hip/hip_bf16.h>

// ---------------------------------------------------------------------------
// Wan self-attention for MI455X (gfx1250, wave32, WMMA + TDM).
// All contractions via v_wmma_f32_16x16x32_f16 (f32 accumulate).
// Matrix-engine bound (~217 GFLOP; ~100MB touched, fits in 192MB L2).
// GEMM B-tiles are DMA'd into LDS by the Tensor Data Mover (double-buffered,
// s_wait_tensorcnt). sched_barrier(0) pins the "stage loads -> single wait ->
// WMMA burst" schedule the compiler otherwise splits.
// ---------------------------------------------------------------------------

typedef __attribute__((ext_vector_type(16))) _Float16 v16h;
typedef __attribute__((ext_vector_type(8)))  _Float16 v8h;
typedef __attribute__((ext_vector_type(4)))  _Float16 v4h;
typedef __attribute__((ext_vector_type(8)))  float    v8f;
typedef __attribute__((ext_vector_type(4)))  unsigned int u32x4;
typedef __attribute__((ext_vector_type(8)))  int          i32x8;
typedef __attribute__((ext_vector_type(4)))  int          i32x4;

#define DIM 1536
#define SEQ 4608
#define NH  12
#define HD  128

#if __has_builtin(__builtin_amdgcn_tensor_load_to_lds)
#define HAVE_TDM 1
#else
#define HAVE_TDM 0
#endif

#if __has_builtin(__builtin_amdgcn_sched_barrier)
#define SCHED_FENCE() __builtin_amdgcn_sched_barrier(0)
#else
#define SCHED_FENCE()
#endif

#if HAVE_TDM
// ---------------------------------------------------------------------------
// Issue a 2-D TDM load: tile_d1 rows x tile_d0 f16 elements, row stride
// stride_d0 elements in global memory, packed contiguously into LDS.
// D# bit layout per CDNA5 ISA ch.8 (group0 128b, group1 256b; groups 2/3
// zero => 2-D tensor). Must be called from a single wave (TDM ignores EXEC).
// ---------------------------------------------------------------------------
__device__ __forceinline__ void tdm_load_2d_f16(
    unsigned lds_off, const _Float16* gptr,
    unsigned tile_d0, unsigned tile_d1, unsigned stride_d0) {
  unsigned long long ga = (unsigned long long)(size_t)gptr;
  u32x4 g0;
  g0[0] = 1u;                                   // count=1, user descriptor
  g0[1] = lds_off;                              // lds_addr (bytes)
  g0[2] = (unsigned)(ga & 0xFFFFFFFFu);         // global_addr[31:0]
  g0[3] = (unsigned)((ga >> 32) & 0x01FFFFFFu)  // global_addr[56:32]
        | 0x80000000u;                          // type=2 ("image")
  i32x8 g1;
  g1[0] = (int)(1u << 16);                      // data_size=1 (2 bytes)
  g1[1] = (int)((tile_d0 & 0xFFFFu) << 16);     // tensor_dim0[15:0] (=tile_d0)
  g1[2] = (int)(((tile_d0 >> 16) & 0xFFFFu) | ((tile_d1 & 0xFFFFu) << 16));
  g1[3] = (int)(((tile_d1 >> 16) & 0xFFFFu) | ((tile_d0 & 0xFFFFu) << 16)); // tile_dim0
  g1[4] = (int)(tile_d1 & 0xFFFFu);             // tile_dim1 (tile_dim2=0)
  g1[5] = (int)stride_d0;                       // tensor_dim0_stride[31:0]
  g1[6] = 0;                                    // stride[47:32]=0, dim1_stride lo=0
  g1[7] = 0;
  i32x4 z4 = {0, 0, 0, 0};
#if __clang_major__ >= 23
  i32x8 z8 = {0, 0, 0, 0, 0, 0, 0, 0};
  __builtin_amdgcn_tensor_load_to_lds(g0, g1, z4, z4, z8, 0);
#else
  __builtin_amdgcn_tensor_load_to_lds(g0, g1, z4, z4, 0);
#endif
}
#endif  // HAVE_TDM

// ---------------------------------------------------------------------------
// f32 -> f16 elementwise convert
// ---------------------------------------------------------------------------
__global__ __launch_bounds__(256) void f32_to_f16_kernel(
    const float* __restrict__ in, _Float16* __restrict__ out, int n) {
  int i = blockIdx.x * 256 + threadIdx.x;
  if (i < n) out[i] = (_Float16)in[i];
}

// ---------------------------------------------------------------------------
// Transpose + convert: w[n][k] f32 -> wt[k][n] f16 (DIM x DIM)
// ---------------------------------------------------------------------------
__global__ __launch_bounds__(256) void transpose_f16_kernel(
    const float* __restrict__ in, _Float16* __restrict__ out) {
  __shared__ float tile[32][33];
  const int bx = blockIdx.x * 32, by = blockIdx.y * 32;
  const int tx = threadIdx.x & 31, ty = threadIdx.x >> 5;  // ty in 0..7
#pragma unroll
  for (int r = 0; r < 32; r += 8)
    tile[ty + r][tx] = in[(size_t)(by + ty + r) * DIM + bx + tx];
  __syncthreads();
#pragma unroll
  for (int r = 0; r < 32; r += 8)
    out[(size_t)(bx + ty + r) * DIM + by + tx] = (_Float16)tile[tx][ty + r];
}

// ---------------------------------------------------------------------------
// WMMA GEMM: C[m][n] = sum_k A[m][k] * Bt[k][n] + bias[n]
// Block: 256 threads = 8 waves (2 M x 4 N); wave tile 16x128; block 32x512.
// B k-tiles (32x512 f16 = 32KB) DMA'd into LDS by TDM, double-buffered so
// the DMA for tile ki+1 overlaps the WMMA burst for tile ki. A tile double-
// buffered in LDS. One barrier per k-step.
// ---------------------------------------------------------------------------
__global__ __launch_bounds__(256) void gemm_wmma_kernel(
    const _Float16* __restrict__ A, const _Float16* __restrict__ Bt,
    const float* __restrict__ bias, _Float16* __restrict__ outh,
    float* __restrict__ outf, int M, int N, int K) {
  __shared__ __attribute__((aligned(32))) _Float16 Bs[2][32][512];
  __shared__ __attribute__((aligned(16))) _Float16 As[2][32][36];

  const int tid  = threadIdx.x;
  const int lane = tid & 31, wave = tid >> 5;
  const int wm = wave & 1, wn = wave >> 1;                  // 2 x 4 wave grid
  const int m0   = blockIdx.x * 32 + wm * 16;
  const int nblk = blockIdx.y * 512;
  const int n0w  = wn * 128;                                // within LDS B tile
  const int lrow  = lane & 15;
  const int kb    = (lane < 16) ? 0 : 8;                    // A frag K sub-offset
  const int kidxB = (lane & 15) + ((lane < 16) ? 0 : 16);   // B frag K row

  const int ldrow = tid >> 3;        // 0..31
  const int ldcol = (tid & 7) * 4;   // 0,4,...,28
  const _Float16* Arow = A + (size_t)(blockIdx.x * 32 + ldrow) * K;

  v8f acc[8] = {};

  // ---- prologue: stage first A chunk + first B tile ----
#if HAVE_TDM
  if (wave == 0)
    tdm_load_2d_f16((unsigned)(size_t)&Bs[0][0][0], Bt + nblk, 512, 32, (unsigned)N);
#else
#pragma unroll
  for (int i = 0; i < 8; ++i) {
    const int c   = tid + i * 256;          // 2048 chunks of 8 halves
    const int row = c >> 6, col8 = (c & 63) * 8;
    *(v8h*)&Bs[0][row][col8] = *(const v8h*)&Bt[(size_t)row * N + nblk + col8];
  }
#endif
  v4h areg = *(const v4h*)&Arow[ldcol];
  *(v4h*)&As[0][ldrow][ldcol] = areg;
#if HAVE_TDM
  if (wave == 0) __builtin_amdgcn_s_wait_tensorcnt(0);
#endif
  __syncthreads();

  const int KSTEPS = K / 32;
  for (int ki = 0; ki < KSTEPS; ++ki) {
    const int cur = ki & 1;
    const int k0  = ki * 32;

    // kick off next B tile DMA + next A chunk load (overlap WMMA burst)
#if HAVE_TDM
    if (wave == 0 && ki + 1 < KSTEPS)
      tdm_load_2d_f16((unsigned)(size_t)&Bs[cur ^ 1][0][0],
                      Bt + (size_t)(k0 + 32) * N + nblk, 512, 32, (unsigned)N);
#endif
    if (ki + 1 < KSTEPS) areg = *(const v4h*)&Arow[k0 + 32 + ldcol];

    // A fragment from LDS
    v16h a;
#pragma unroll
    for (int j = 0; j < 8; ++j) {
      a[j]     = As[cur][wm * 16 + lrow][kb + j];
      a[j + 8] = As[cur][wm * 16 + lrow][kb + 16 + j];
    }

    // stage all 8 B fragments from the LDS tile ...
    v16h bf[8];
#pragma unroll
    for (int t = 0; t < 8; ++t)
#pragma unroll
      for (int j = 0; j < 8; ++j) {
        bf[t][j]     = Bs[cur][kb + j][n0w + t * 16 + lrow];
        bf[t][j + 8] = Bs[cur][kb + 16 + j][n0w + t * 16 + lrow];
      }
    // ... then a pinned burst of 8 WMMAs (one ds wait, back-to-back XDL ops)
    SCHED_FENCE();
#pragma unroll
    for (int t = 0; t < 8; ++t)
      acc[t] = __builtin_amdgcn_wmma_f32_16x16x32_f16(
          false, a, false, bf[t], (short)0, acc[t], false, false);
    SCHED_FENCE();

    if (ki + 1 < KSTEPS) {
      *(v4h*)&As[cur ^ 1][ldrow][ldcol] = areg;
#if !HAVE_TDM
#pragma unroll
      for (int i = 0; i < 8; ++i) {
        const int c   = tid + i * 256;
        const int row = c >> 6, col8 = (c & 63) * 8;
        *(v8h*)&Bs[cur ^ 1][row][col8] =
            *(const v8h*)&Bt[(size_t)(k0 + 32 + row) * N + nblk + col8];
      }
#endif
    }
#if HAVE_TDM
    if (wave == 0) __builtin_amdgcn_s_wait_tensorcnt(0);
#endif
    __syncthreads();
  }

#pragma unroll
  for (int t = 0; t < 8; ++t) {
#pragma unroll
    for (int r = 0; r < 8; ++r) {
      const int row = m0 + r + ((lane < 16) ? 0 : 8);
      const int col = nblk + n0w + t * 16 + (lane & 15);
      const float val = acc[t][r] + bias[col];
      if (outf) outf[(size_t)row * N + col] = val;
      else      outh[(size_t)row * N + col] = (_Float16)val;
    }
  }
}

// ---------------------------------------------------------------------------
// RMSNorm (full 1536-dim row, fp32 stats) + RoPE (interleaved pairs).
// mode 0: q, normalized+rotated in place.
// mode 1: k, normalized+rotated into transposed kt[h*128+d][s] buffer.
// ---------------------------------------------------------------------------
__global__ __launch_bounds__(256) void normrope_kernel(
    _Float16* __restrict__ buf, const float* __restrict__ g,
    const float* __restrict__ freqs, _Float16* __restrict__ kt, int mode) {
  __shared__ float red[256];
  __shared__ float s_rs;
  const int s = blockIdx.x;
  const int tid = threadIdx.x;
  _Float16* row = buf + (size_t)s * DIM;

  float local = 0.f;
  for (int c = tid; c < DIM; c += 256) {
    float v = (float)row[c];
    local += v * v;
  }
  red[tid] = local;
  __syncthreads();
  for (int off = 128; off > 0; off >>= 1) {
    if (tid < off) red[tid] += red[tid + off];
    __syncthreads();
  }
  if (tid == 0) s_rs = rsqrtf(red[0] * (1.0f / DIM) + 1e-6f);
  __syncthreads();
  const float rs = s_rs;

  // grid position (F=8, H=24, W=24)
  const int fidx = s / 576;
  const int rem  = s - fidx * 576;
  const int hidx = rem / 24;
  const int widx = rem - hidx * 24;

  for (int p = tid; p < DIM / 2; p += 256) {
    const int i  = p & 63;           // rope angle column within head
    const int c0 = 2 * p, c1 = c0 + 1;
    const int frow = (i < 22) ? fidx : ((i < 43) ? hidx : widx);
    const float ang = freqs[frow * 64 + i];
    const float cs = __cosf(ang), sn = __sinf(ang);
    const float x0 = (float)row[c0] * rs * g[c0];
    const float x1 = (float)row[c1] * rs * g[c1];
    const float re = x0 * cs - x1 * sn;
    const float im = x0 * sn + x1 * cs;
    if (mode == 0) {
      row[c0] = (_Float16)re;
      row[c1] = (_Float16)im;
    } else {
      kt[(size_t)c0 * SEQ + s] = (_Float16)re;
      kt[(size_t)c1 * SEQ + s] = (_Float16)im;
    }
  }
}

// ---------------------------------------------------------------------------
// Flash attention. Block: 8 waves; each wave owns one 16-row Q tile of one
// head and streams 32-wide key tiles: S = Q Kt (WMMA burst), online softmax
// in per-wave LDS, O += P V (WMMA burst, P transposed through LDS).
// ---------------------------------------------------------------------------
__global__ __launch_bounds__(256) void attn_kernel(
    const _Float16* __restrict__ q, const _Float16* __restrict__ kt,
    const _Float16* __restrict__ v, _Float16* __restrict__ o) {
  __shared__ float     Sc[8][16][33];
  __shared__ _Float16  Pl[8][16][40];
  __shared__ float     mi[8][16], li[8][16], corr[8][16], invl[8][16];

  const int tid  = threadIdx.x;
  const int lane = tid & 31, w = tid >> 5;
  const int h  = blockIdx.y;
  const int m0 = (blockIdx.x * 8 + w) * 16;
  const int lrow  = lane & 15;
  const int kb    = (lane < 16) ? 0 : 8;
  const int kidxB = (lane & 15) + ((lane < 16) ? 0 : 16);
  const float scale = 0.08838834764831845f;  // 1/sqrt(128)

  // preload Q fragments (head_dim 128 = 4 WMMA K-steps of 32)
  v16h qa[4];
  {
    const _Float16* qrow = q + (size_t)(m0 + lrow) * DIM + h * HD;
#pragma unroll
    for (int ks = 0; ks < 4; ++ks)
#pragma unroll
      for (int j = 0; j < 8; ++j) {
        qa[ks][j]     = qrow[ks * 32 + kb + j];
        qa[ks][j + 8] = qrow[ks * 32 + kb + 16 + j];
      }
  }

  v8f oacc[8] = {};
  if (lane < 16) { mi[w][lane] = -1e30f; li[w][lane] = 0.f; }
  __syncthreads();

  for (int n0k = 0; n0k < SEQ; n0k += 32) {
    // ---- stage all 8 K fragments, then pinned burst of 8 score WMMAs ----
    v16h kf[8];
#pragma unroll
    for (int ks = 0; ks < 4; ++ks) {
      const _Float16* kbase =
          kt + (size_t)(h * HD + ks * 32) * SEQ + n0k + lrow;
#pragma unroll
      for (int j = 0; j < 8; ++j) {
        kf[2 * ks][j]         = kbase[(size_t)(kb + j) * SEQ];
        kf[2 * ks][j + 8]     = kbase[(size_t)(kb + 16 + j) * SEQ];
        kf[2 * ks + 1][j]     = kbase[(size_t)(kb + j) * SEQ + 16];
        kf[2 * ks + 1][j + 8] = kbase[(size_t)(kb + 16 + j) * SEQ + 16];
      }
      if (n0k + 32 < SEQ) __builtin_prefetch(kbase + 32, 0, 0);
    }
    SCHED_FENCE();
    v8f s0 = {}, s1 = {};
#pragma unroll
    for (int ks = 0; ks < 4; ++ks) {
      s0 = __builtin_amdgcn_wmma_f32_16x16x32_f16(
          false, qa[ks], false, kf[2 * ks], (short)0, s0, false, false);
      s1 = __builtin_amdgcn_wmma_f32_16x16x32_f16(
          false, qa[ks], false, kf[2 * ks + 1], (short)0, s1, false, false);
    }
    SCHED_FENCE();
#pragma unroll
    for (int r = 0; r < 8; ++r) {
      const int row = r + ((lane < 16) ? 0 : 8);
      Sc[w][row][lane & 15]        = s0[r] * scale;
      Sc[w][row][16 + (lane & 15)] = s1[r] * scale;
    }
    __syncthreads();

    // ---- online softmax (lane r < 16 owns row r) ----
    if (lane < 16) {
      float mold = mi[w][lane];
      float mnew = mold;
#pragma unroll 4
      for (int c = 0; c < 32; ++c) mnew = fmaxf(mnew, Sc[w][lane][c]);
      const float cr = __expf(mold - mnew);
      float l = li[w][lane] * cr;
#pragma unroll 4
      for (int c = 0; c < 32; ++c) {
        const float p = __expf(Sc[w][lane][c] - mnew);
        l += p;
        Pl[w][lane][c] = (_Float16)p;
      }
      mi[w][lane] = mnew;
      li[w][lane] = l;
      corr[w][lane] = cr;
    }
    __syncthreads();

    // ---- rescale accumulator rows by correction factor ----
#pragma unroll
    for (int r = 0; r < 8; ++r) {
      const float cr = corr[w][r + ((lane < 16) ? 0 : 8)];
#pragma unroll
      for (int t = 0; t < 8; ++t) oacc[t][r] *= cr;
    }

    // ---- P fragment + staged V fragments, then pinned burst of 8 WMMAs ----
    v16h pa;
#pragma unroll
    for (int j = 0; j < 8; ++j) {
      pa[j]     = Pl[w][lrow][kb + j];
      pa[j + 8] = Pl[w][lrow][kb + 16 + j];
    }
    const _Float16* vbase = v + (size_t)n0k * DIM + h * HD + lrow;
    v16h vf[8];
#pragma unroll
    for (int t = 0; t < 8; ++t)
#pragma unroll
      for (int j = 0; j < 8; ++j) {
        vf[t][j]     = vbase[(size_t)(kb + j) * DIM + t * 16];
        vf[t][j + 8] = vbase[(size_t)(kb + 16 + j) * DIM + t * 16];
      }
    SCHED_FENCE();
#pragma unroll
    for (int t = 0; t < 8; ++t)
      oacc[t] = __builtin_amdgcn_wmma_f32_16x16x32_f16(
          false, pa, false, vf[t], (short)0, oacc[t], false, false);
    SCHED_FENCE();
    __syncthreads();
  }

  if (lane < 16) invl[w][lane] = 1.0f / li[w][lane];
  __syncthreads();

#pragma unroll
  for (int t = 0; t < 8; ++t) {
#pragma unroll
    for (int r = 0; r < 8; ++r) {
      const int row = m0 + r + ((lane < 16) ? 0 : 8);
      const int col = h * HD + t * 16 + (lane & 15);
      const float val = oacc[t][r] * invl[w][r + ((lane < 16) ? 0 : 8)];
      o[(size_t)row * DIM + col] = (_Float16)val;
    }
  }
}

// ---------------------------------------------------------------------------
// Host-side launcher
// ---------------------------------------------------------------------------
extern "C" void kernel_launch(void* const* d_in, const int* in_sizes, int n_in,
                              void* d_out, int out_size, void* d_ws, size_t ws_size,
                              hipStream_t stream) {
  (void)in_sizes; (void)n_in; (void)out_size; (void)ws_size;
  const float* x     = (const float*)d_in[0];
  const float* wq    = (const float*)d_in[1];
  const float* bq    = (const float*)d_in[2];
  const float* wk    = (const float*)d_in[3];
  const float* bk    = (const float*)d_in[4];
  const float* wv    = (const float*)d_in[5];
  const float* bv    = (const float*)d_in[6];
  const float* wo    = (const float*)d_in[7];
  const float* bo    = (const float*)d_in[8];
  const float* gq    = (const float*)d_in[9];
  const float* gk    = (const float*)d_in[10];
  const float* freqs = (const float*)d_in[11];
  float* out = (float*)d_out;

  char* ws = (char*)d_ws;
  size_t off = 0;
  auto alloc = [&](size_t bytes) -> char* {
    char* p = ws + off;
    off += (bytes + 255) & ~(size_t)255;
    return p;
  };
  const size_t act_bytes = (size_t)SEQ * DIM * sizeof(_Float16);
  const size_t w_bytes   = (size_t)DIM * DIM * sizeof(_Float16);
  _Float16* xh   = (_Float16*)alloc(act_bytes);  // x in f16; reused as attn out
  _Float16* wqT  = (_Float16*)alloc(w_bytes);
  _Float16* wkT  = (_Float16*)alloc(w_bytes);
  _Float16* wvT  = (_Float16*)alloc(w_bytes);
  _Float16* woT  = (_Float16*)alloc(w_bytes);
  _Float16* qbuf = (_Float16*)alloc(act_bytes);
  _Float16* kbuf = (_Float16*)alloc(act_bytes);
  _Float16* vbuf = (_Float16*)alloc(act_bytes);
  _Float16* ktb  = (_Float16*)alloc(act_bytes);  // k transposed: [h*128+d][s]
  _Float16* attn = xh;                           // reuse xh after QKV gemms

  // 1) convert x to f16
  {
    const int n = SEQ * DIM;
    f32_to_f16_kernel<<<(n + 255) / 256, 256, 0, stream>>>(x, xh, n);
  }
  // 2) transpose+convert weights: w[n][k] -> wT[k][n] f16
  {
    dim3 g(DIM / 32, DIM / 32);
    transpose_f16_kernel<<<g, 256, 0, stream>>>(wq, wqT);
    transpose_f16_kernel<<<g, 256, 0, stream>>>(wk, wkT);
    transpose_f16_kernel<<<g, 256, 0, stream>>>(wv, wvT);
    transpose_f16_kernel<<<g, 256, 0, stream>>>(wo, woT);
  }
  // 3) QKV projections (WMMA + TDM)
  {
    dim3 g(SEQ / 32, DIM / 512);
    gemm_wmma_kernel<<<g, 256, 0, stream>>>(xh, wqT, bq, qbuf, nullptr, SEQ, DIM, DIM);
    gemm_wmma_kernel<<<g, 256, 0, stream>>>(xh, wkT, bk, kbuf, nullptr, SEQ, DIM, DIM);
    gemm_wmma_kernel<<<g, 256, 0, stream>>>(xh, wvT, bv, vbuf, nullptr, SEQ, DIM, DIM);
  }
  // 4) RMSNorm + RoPE: q in place, k into transposed kt buffer
  normrope_kernel<<<SEQ, 256, 0, stream>>>(qbuf, gq, freqs, nullptr, 0);
  normrope_kernel<<<SEQ, 256, 0, stream>>>(kbuf, gk, freqs, ktb, 1);
  // 5) flash attention (WMMA), writes f16 attn (= xh buffer)
  {
    dim3 g(SEQ / 16 / 8, NH);
    attn_kernel<<<g, 256, 0, stream>>>(qbuf, ktb, vbuf, attn);
  }
  // 6) output projection -> f32 d_out (WMMA + TDM)
  {
    dim3 g(SEQ / 32, DIM / 512);
    gemm_wmma_kernel<<<g, 256, 0, stream>>>(attn, woT, bo, nullptr, out, SEQ, DIM, DIM);
  }
}
